// CFormerAdapter_12652973654480
// MI455X (gfx1250) — matrix operand order repaired and checked by reading, not compiled, not run on hardware
//
#include <hip/hip_runtime.h>

// ---------------- problem constants ----------------
#define BATCH 8
#define TLEN  1500
#define TPAD  1504      // 47 * 32
#define HDIM  1280
#define HM1   1279
#define ODIM  4096
#define MTOK  375
#define MPAD  384       // 3 * 128

// ---------------- WMMA / TDM types ----------------
typedef __attribute__((ext_vector_type(16))) __bf16 v16bf;
typedef __attribute__((ext_vector_type(8)))  float  v8f;
typedef __attribute__((ext_vector_type(4)))  unsigned int u32x4;
typedef __attribute__((ext_vector_type(8)))  int          i32x8;
typedef __attribute__((ext_vector_type(4)))  int          i32x4;

union FragBF { v16bf v; unsigned short u[16]; };

#ifndef __has_builtin
#define __has_builtin(x) 0
#endif
#if defined(__HIP_DEVICE_COMPILE__) && __has_builtin(__builtin_amdgcn_tensor_load_to_lds)
#define HAVE_TDM 1
#else
#define HAVE_TDM 0
#endif

__device__ __forceinline__ unsigned short f2bf(float f) {
    union { float f; unsigned int u; } x; x.f = f;
    unsigned int u = x.u;
    unsigned int r = u + 0x7FFFu + ((u >> 16) & 1u);   // round-to-nearest-even
    return (unsigned short)(r >> 16);
}
__device__ __forceinline__ float bf2f(unsigned short h) {
    union { unsigned int u; float f; } x; x.u = (unsigned int)h << 16;
    return x.f;
}

// ---------------- alphas: sigmoid, batch-sum, rescale; writes pred tail ----------------
__global__ __launch_bounds__(256)
void alphas_kernel(const float* __restrict__ audio, const int* __restrict__ num_tokens,
                   float* __restrict__ alphas, float* __restrict__ out) {
    const int b = blockIdx.x, tid = threadIdx.x;
    __shared__ float red[256];
    float s = 0.f;
    for (int t = tid; t < TLEN; t += 256) {
        float x = audio[((long)b * TLEN + t) * HDIM + (HDIM - 1)];
        float a = 1.f / (1.f + __expf(-x));
        alphas[b * TLEN + t] = a;
        s += a;
    }
    red[tid] = s; __syncthreads();
    for (int off = 128; off > 0; off >>= 1) {
        if (tid < off) red[tid] += red[tid + off];
        __syncthreads();
    }
    float total = red[0];
    if (tid == 0) out[(long)BATCH * MTOK * ODIM + b] = total;  // pred_num_tokens
    float scale = (float)num_tokens[b] / total;
    for (int t = tid; t < TLEN; t += 256) alphas[b * TLEN + t] *= scale;
}

// ---------------- zero fill (float4) ----------------
__global__ __launch_bounds__(256)
void zero_f4(float4* __restrict__ p, long n4) {
    long i = (long)blockIdx.x * 256 + threadIdx.x;
    if (i < n4) p[i] = make_float4(0.f, 0.f, 0.f, 0.f);
}

// ------- hidden -> bf16 TRANSPOSED [B, HDIM, TPAD]; (d>=HM1 or t>=TLEN) = 0 -------
__global__ __launch_bounds__(256)
void conv_hidden_t(const float* __restrict__ audio, unsigned short* __restrict__ o) {
    long i = (long)blockIdx.x * 256 + threadIdx.x;
    const long n = (long)BATCH * HDIM * TPAD;
    if (i >= n) return;
    int t = (int)(i % TPAD);
    long r = i / TPAD;
    int d = (int)(r % HDIM);
    int b = (int)(r / HDIM);
    float v = 0.f;
    if (t < TLEN && d < HM1) v = audio[((long)b * TLEN + t) * HDIM + d];
    o[i] = f2bf(v);
}

// ------- cif_w [HM1, HDIM] -> bf16 TRANSPOSED [HDIM, HDIM] (k >= HM1 zero) -------
__global__ __launch_bounds__(256)
void conv_cifw_t(const float* __restrict__ w, unsigned short* __restrict__ o) {
    long i = (long)blockIdx.x * 256 + threadIdx.x;
    if (i >= (long)HDIM * HDIM) return;
    int k = (int)(i % HDIM);
    int nn = (int)(i / HDIM);
    o[i] = f2bf((k < HM1) ? w[(long)k * HDIM + nn] : 0.f);
}

// ------- text_w [HDIM, ODIM] -> bf16 TRANSPOSED [ODIM, HDIM] -------
__global__ __launch_bounds__(256)
void conv_textw_t(const float* __restrict__ w, unsigned short* __restrict__ o) {
    long i = (long)blockIdx.x * 256 + threadIdx.x;
    if (i >= (long)ODIM * HDIM) return;
    int k = (int)(i % HDIM);
    int nn = (int)(i / HDIM);
    o[i] = f2bf(w[(long)k * ODIM + nn]);
}

// ---------------- CIF scan: one wave32, lane b = batch b; writes bf16 weights ------
__global__ __launch_bounds__(32)
void cif_scan(const float* __restrict__ alphas, const int* __restrict__ num_tokens,
              unsigned short* __restrict__ W) {
    const int lane = threadIdx.x;
    const bool real = lane < BATCH;
    const int b = real ? lane : 0;
    const int nm1 = real ? (num_tokens[b] - 1) : 0;
    unsigned short* wrow = W + (long)b * MPAD * TPAD;

    __shared__ float buf[BATCH * 64];   // 64-step alpha chunks for all batches

    float integrate = 0.f;
    int token_index = 0;

    for (int tc = 0; tc < TLEN; tc += 64) {
        __syncthreads();
        for (int j = 0; j < 16; ++j) {
            int idx = lane + j * 32;            // 512 entries
            int bb = idx >> 6, tt = idx & 63;
            float v = 0.f;
            if (tc + tt < TLEN) v = alphas[bb * TLEN + tc + tt];
            buf[idx] = v;
        }
        __syncthreads();

        const int tend = min(64, TLEN - tc);
        for (int ti = 0; ti < tend; ++ti) {
            const int t = tc + ti;
            float alpha = real ? buf[(lane << 6) + ti] : 0.f;
            float alpha_needed = 1.f - integrate;
            integrate += alpha;
            bool ready = integrate >= 1.f;
            float remainds = 0.f;
            bool active = true;                 // do-while: first body iteration always runs
            #pragma unroll
            for (int it = 0; it < 3; ++it) {
                float n_int = ready ? (integrate - 1.f) : integrate;
                float a_int = ready ? alpha_needed : alpha;
                float n_rem = alpha - a_int;
                int   n_tok = min(token_index + (ready ? 1 : 0), nm1);
                bool  n_ready = n_int >= 1.f;
                if (active) {
                    if (real) wrow[(long)token_index * TPAD + t] = f2bf(a_int);  // col.set
                    integrate = n_int;
                    remainds = n_rem;
                    token_index = n_tok;
                    alpha = n_rem;
                    alpha_needed = 1.f;
                    ready = n_ready;
                } else {
                    ready = false;
                }
                active = active && __any((int)ready);   // cross-batch any(), as in reference
            }
            if (real && t < TLEN - 1) {
                long idx = (long)token_index * TPAD + t;
                wrow[idx] = f2bf(bf2f(wrow[idx]) + remainds);
            }
        }
    }
}

// ---------------- bf16 WMMA GEMM: C = A(bf16)[M,K] * Bt(bf16)[N,K]^T + bias ----------
// block tile 128x128, BK=32; 8 waves as 4(M) x 2(N); each wave 32x64 -> 8 wmma / K-step.
// Both tiles DMA'd by the Tensor Data Mover into double-buffered LDS (padded strides via
// D# pad_interval/pad_amount); DMA of tile i+1 overlaps the WMMAs of tile i.
#define BMT 128
#define BNT 128
#define BKT 32
#define SAS 40   // A LDS row stride (halves): 32 + TDM pad 4 DW
#define SBS 48   // B LDS row stride (halves): 32 + TDM pad 8 DW

#if HAVE_TDM
__device__ __forceinline__ void tdm_tile_load(unsigned long long ga, unsigned int lds,
                                              int K, int rows_total, int row_stride,
                                              unsigned int pad_amt_code) {
    u32x4 g0;
    g0.x = 1u;                                          // count=1, user descriptor
    g0.y = lds;                                         // lds_addr
    g0.z = (unsigned int)ga;                            // global_addr[31:0]
    g0.w = (unsigned int)((ga >> 32) & 0x1FFFFFFull) | 0x80000000u; // addr hi | type=2
    i32x8 g1;
    g1[0] = (int)(0x00010000u |                         // data_size = 2B
                  (1u << 20) |                          // pad_enable
                  (3u << 22) |                          // pad_interval: 16 DWORDs (= 32 halves)
                  (pad_amt_code << 25));                // pad_amount DWORDs (code = n-1)
    g1[1] = (int)(((unsigned int)K & 0xFFFFu) << 16);   // tensor_dim0 lo16
    g1[2] = (int)(((unsigned int)K >> 16) |
                  (((unsigned int)rows_total & 0xFFFFu) << 16)); // dim0 hi | dim1 lo
    g1[3] = (int)((((unsigned int)rows_total >> 16) & 0xFFFFu) | (32u << 16)); // dim1 hi | tile_dim0=32
    g1[4] = (int)128u;                                  // tile_dim1 = 128, tile_dim2 = 0
    g1[5] = (int)(unsigned int)row_stride;              // tensor_dim0_stride lo32
    g1[6] = 0; g1[7] = 0;
    i32x4 gz = {0, 0, 0, 0};
#if __clang_major__ >= 23
    i32x8 gz8 = {0,0,0,0,0,0,0,0};
    __builtin_amdgcn_tensor_load_to_lds(g0, g1, gz, gz, gz8, 0);
#else
    __builtin_amdgcn_tensor_load_to_lds(g0, g1, gz, gz, 0);
#endif
}
#endif

template <bool STORE_BF16>
__global__ __launch_bounds__(256)
void gemm_bf16_wmma(const unsigned short* __restrict__ A, const unsigned short* __restrict__ Bt,
                    const float* __restrict__ bias, void* __restrict__ Cv,
                    int M, int K, int N, int lda, int ldbt, int ldc,
                    long strideAb, long strideBb, long strideCb, int m_limit) {
    __shared__ unsigned short As[2][BMT * SAS];
    __shared__ unsigned short Bs[2][BNT * SBS];

    const int tid = threadIdx.x;
    const int bz  = blockIdx.z;
    const int blockM = blockIdx.y * BMT;
    const int blockN = blockIdx.x * BNT;

    const unsigned short* Ab  = A  + (long)bz * strideAb;
    const unsigned short* Btb = Bt + (long)bz * strideBb;

    const int lane  = tid & 31;
    const int wv    = tid >> 5;
    const int l16   = lane & 15;
    const int khalf = lane >> 4;
    const int wm    = wv & 3;       // M sub-tile: 32*wm
    const int wn    = wv >> 2;      // N sub-tile: 64*wn

    v8f acc[2][4];
    #pragma unroll
    for (int i = 0; i < 2; ++i)
        #pragma unroll
        for (int j = 0; j < 4; ++j)
            acc[i][j] = (v8f){0.f,0.f,0.f,0.f,0.f,0.f,0.f,0.f};

#if HAVE_TDM
    const unsigned long long gbaseA =
        (unsigned long long)(size_t)(const void*)(Ab + (long)blockM * lda);
    const unsigned long long gbaseB =
        (unsigned long long)(size_t)(const void*)(Btb + (long)blockN * ldbt);
#endif

    // stage tile `it` (K offset it*BKT) into LDS buffer `bufi`
    auto stage = [&](int it, int bufi) {
#if HAVE_TDM
        const int k0 = it * BKT;
        if (wv == 0)            // B tile: 128 rows x 32 halves, pad to 48-half stride
            tdm_tile_load(gbaseB + (unsigned long long)k0 * 2ull,
                          (unsigned int)(size_t)(void*)&Bs[bufi][0],
                          K, N, ldbt, 7u);
        else if (wv == 1)       // A tile: 128 rows x 32 halves, pad to 40-half stride
            tdm_tile_load(gbaseA + (unsigned long long)k0 * 2ull,
                          (unsigned int)(size_t)(void*)&As[bufi][0],
                          K, M, lda, 3u);
#else
        const int k0 = it * BKT;
        #pragma unroll
        for (int i = 0; i < 2; ++i) {                 // B: 512 x uint4 (8 halves)
            int idx = i * 256 + tid;
            int row = idx >> 2;
            int k8  = (idx & 3) * 8;
            uint4 v = *(const uint4*)&Btb[(long)(blockN + row) * ldbt + k0 + k8];
            *(uint4*)((char*)&Bs[bufi][0] + ((long)row * SBS + k8) * 2) = v;
        }
        #pragma unroll
        for (int i = 0; i < 2; ++i) {                 // A: 512 x uint4 (8 halves)
            int idx = i * 256 + tid;
            int row = idx >> 2;
            int k8  = (idx & 3) * 8;
            uint4 v = *(const uint4*)&Ab[(long)(blockM + row) * lda + k0 + k8];
            *(uint4*)((char*)&As[bufi][0] + ((long)row * SAS + k8) * 2) = v;
        }
#endif
    };

    const int niter = K / BKT;
    stage(0, 0);

    for (int it = 0; it < niter; ++it) {
        const int cur = it & 1;
#if HAVE_TDM
        if (wv < 2) __builtin_amdgcn_s_wait_tensorcnt(0);  // tile `it` landed in LDS
#endif
        __syncthreads();
        if (it + 1 < niter) stage(it + 1, cur ^ 1);        // DMA overlaps WMMAs below

        // A fragments (16x32, contiguous k order, same as B): m = lane&15; k = 16*khalf + e
        FragBF a[2];
        #pragma unroll
        for (int mf = 0; mf < 2; ++mf)
            #pragma unroll
            for (int e = 0; e < 16; ++e) {
                int k = (khalf << 4) + e;
                a[mf].u[e] = As[cur][(wm * 32 + mf * 16 + l16) * SAS + k];
            }
        // B fragments (32x16): n = lane&15, k = 16*khalf + e (K-contiguous in LDS)
        #pragma unroll
        for (int nf = 0; nf < 4; ++nf) {
            FragBF bb;
            const int nrow = wn * 64 + nf * 16 + l16;
            const int kb = khalf << 4;
            #pragma unroll
            for (int e = 0; e < 16; ++e) bb.u[e] = Bs[cur][nrow * SBS + kb + e];
            acc[0][nf] = __builtin_amdgcn_wmma_f32_16x16x32_bf16(
                false, a[0].v, false, bb.v, (short)0, acc[0][nf], false, false);
            acc[1][nf] = __builtin_amdgcn_wmma_f32_16x16x32_bf16(
                false, a[1].v, false, bb.v, (short)0, acc[1][nf], false, false);
        }
        __syncthreads();
    }

    // store: C/D layout: VGPR r -> M = r (lanes 0-15) / r+8 (lanes 16-31), N = lane&15
    #pragma unroll
    for (int mf = 0; mf < 2; ++mf)
        #pragma unroll
        for (int nf = 0; nf < 4; ++nf) {
            int n = blockN + wn * 64 + nf * 16 + l16;
            float bv = bias ? bias[n] : 0.f;
            #pragma unroll
            for (int r = 0; r < 8; ++r) {
                int m = blockM + wm * 32 + mf * 16 + r + (khalf << 3);
                if (m < m_limit) {
                    float v = acc[mf][nf][r] + bv;
                    if (STORE_BF16)
                        ((unsigned short*)Cv)[(long)bz * strideCb + (long)m * ldc + n] = f2bf(v);
                    else
                        ((float*)Cv)[(long)bz * strideCb + (long)m * ldc + n] = v;
                }
            }
        }
}

// ---------------- RMSNorm rows of h2 (bf16) [B*MPAD, HDIM] in place ----------------
__global__ __launch_bounds__(256)
void rmsnorm_bf16(unsigned short* __restrict__ h, const float* __restrict__ rms_w) {
    const long row = blockIdx.x;
    unsigned short* p = h + row * HDIM;
    const int tid = threadIdx.x;
    __shared__ float red[256];
    float s = 0.f;
    for (int d = tid; d < HDIM; d += 256) { float v = bf2f(p[d]); s += v * v; }
    red[tid] = s; __syncthreads();
    for (int off = 128; off > 0; off >>= 1) {
        if (tid < off) red[tid] += red[tid + off];
        __syncthreads();
    }
    float inv = rsqrtf(red[0] / (float)HDIM + 1e-6f);
    for (int d = tid; d < HDIM; d += 256) p[d] = f2bf(bf2f(p[d]) * inv * rms_w[d]);
}

// ---------------- launch ----------------
extern "C" void kernel_launch(void* const* d_in, const int* in_sizes, int n_in,
                              void* d_out, int out_size, void* d_ws, size_t ws_size,
                              hipStream_t stream) {
    (void)in_sizes; (void)n_in; (void)out_size; (void)ws_size;
    const float* audio      = (const float*)d_in[0];
    const int*   num_tokens = (const int*)d_in[1];
    const float* rms_w      = (const float*)d_in[2];
    const float* cif_w      = (const float*)d_in[3];
    const float* cif_b      = (const float*)d_in[4];
    const float* text_w     = (const float*)d_in[5];
    const float* text_b     = (const float*)d_in[6];
    float* out = (float*)d_out;

    char* ws = (char*)d_ws;
    size_t off = 0;
    auto alloc = [&](size_t bytes) -> void* {
        void* p = ws + off;
        off = (off + bytes + 255) & ~(size_t)255;
        return p;
    };
    float*          alphas  = (float*)alloc((size_t)BATCH * TLEN * 4);
    unsigned short* weights = (unsigned short*)alloc((size_t)BATCH * MPAD * TPAD * 2);
    unsigned short* hid_t   = (unsigned short*)alloc((size_t)BATCH * HDIM * TPAD * 2);
    unsigned short* cifw_t  = (unsigned short*)alloc((size_t)HDIM * HDIM * 2);
    unsigned short* textw_t = (unsigned short*)alloc((size_t)ODIM * HDIM * 2);
    unsigned short* h1      = (unsigned short*)alloc((size_t)BATCH * MPAD * HDIM * 2);
    unsigned short* h2      = (unsigned short*)alloc((size_t)BATCH * MPAD * HDIM * 2);

    // 1) alphas + pred_num_tokens
    alphas_kernel<<<BATCH, 256, 0, stream>>>(audio, num_tokens, alphas, out);

    // 2) zero the padded bf16 weights buffer
    {
        long n4 = (long)BATCH * MPAD * TPAD * 2 / 16;
        zero_f4<<<(unsigned)((n4 + 255) / 256), 256, 0, stream>>>((float4*)weights, n4);
    }
    // 3) bf16 conversions, B operands stored transposed [N, K]
    {
        long n = (long)BATCH * HDIM * TPAD;
        conv_hidden_t<<<(unsigned)((n + 255) / 256), 256, 0, stream>>>(audio, hid_t);
        long nc = (long)HDIM * HDIM;
        conv_cifw_t<<<(unsigned)((nc + 255) / 256), 256, 0, stream>>>(cif_w, cifw_t);
        long nt = (long)ODIM * HDIM;
        conv_textw_t<<<(unsigned)((nt + 255) / 256), 256, 0, stream>>>(text_w, textw_t);
    }
    // 4) sequential CIF scan (one wave32; lanes 0-7 = batches)
    cif_scan<<<1, 32, 0, stream>>>(alphas, num_tokens, weights);

    // 5) h1 = weights @ hidden   [MPAD x TPAD] x [TPAD x HDIM], per batch -> bf16
    gemm_bf16_wmma<true><<<dim3(HDIM / BNT, MPAD / BMT, BATCH), 256, 0, stream>>>(
        weights, hid_t, nullptr, h1,
        MPAD, TPAD, HDIM, TPAD, TPAD, HDIM,
        (long)MPAD * TPAD, (long)HDIM * TPAD, (long)MPAD * HDIM, MPAD);

    // 6) h2 = h1 @ cif_w + cif_b   [MPAD x HDIM] x [HDIM x HDIM] -> bf16
    gemm_bf16_wmma<true><<<dim3(HDIM / BNT, MPAD / BMT, BATCH), 256, 0, stream>>>(
        h1, cifw_t, cif_b, h2,
        MPAD, HDIM, HDIM, HDIM, HDIM, HDIM,
        (long)MPAD * HDIM, 0L, (long)MPAD * HDIM, MPAD);

    // 7) RMSNorm rows in place (bf16)
    rmsnorm_bf16<<<BATCH * MPAD, 256, 0, stream>>>(h2, rms_w);

    // 8) out = h2 @ text_w + text_b -> fp32, only m < 375 rows packed [B,375,ODIM]
    gemm_bf16_wmma<false><<<dim3(ODIM / BNT, MPAD / BMT, BATCH), 256, 0, stream>>>(
        h2, textw_t, text_b, out,
        MPAD, HDIM, ODIM, HDIM, HDIM, ODIM,
        (long)MPAD * HDIM, 0L, (long)MTOK * ODIM, MTOK);
}
